// SimpleAtomInteraction_6932077216273
// MI455X (gfx1250) — matrix operand order repaired and checked by reading, not compiled, not run on hardware
//
#include <hip/hip_runtime.h>

typedef __attribute__((ext_vector_type(16))) _Float16 v16h;
typedef __attribute__((ext_vector_type(8)))  _Float16 v8h;
typedef __attribute__((ext_vector_type(8)))  float    v8f;

#define Bsz   8
#define Adim  512
#define Ndim  64
#define Fdim  128
#define Gdim  50
#define NROWS 4096           // B*A
#define LDA1  72             // f16 stride, K-pad for stage-1 A (fij), mult of 8
#define LDB   144            // f16 stride for col-major B in LDS, mult of 16
#define LDA2  136            // f16 stride for stage-2 A, mult of 8
#define LDY   132            // f32 stride for gathered y tile

__device__ __forceinline__ float sspf(float x) {
  // softplus(x) - log(2), numerically stable
  float t = fmaxf(x, 0.0f) + log1pf(expf(-fabsf(x)));
  return t - 0.69314718055994531f;
}
__device__ __forceinline__ float cutoff_w(float r) {
  return (r < 5.0f) ? (0.5f * (cosf(r * 0.62831853071795864f) + 1.0f)) : 0.0f;
}

// A fragment (16x32 f16): lane m=lane&15, h=lane>>4; elem e -> K=k0+16h+e
// (same k order as frag_b; two b128 DS loads)
__device__ __forceinline__ v16h frag_a(const _Float16* As, int lda, int row0, int k0) {
  int lane = threadIdx.x & 31;
  int m = lane & 15, h = lane >> 4;
  const _Float16* p = As + (row0 + m) * lda + k0 + h * 16;
  v8h lo = *(const v8h*)(p);
  v8h hi = *(const v8h*)(p + 8);
  v16h f;
#pragma unroll
  for (int i = 0; i < 8; ++i) { f[i] = lo[i]; f[i + 8] = hi[i]; }
  return f;
}

// B fragment (32x16 f16): lane n=lane&15, kh=lane>>4; elem e -> K=k0+16*kh+e.
// BsT is column-major (BsT[n*LDB + k]) so this is one aligned 32B run.
__device__ __forceinline__ v16h frag_b(const _Float16* BsT, int col0, int k0) {
  int lane = threadIdx.x & 31;
  int n = lane & 15, kh = lane >> 4;
  return *(const v16h*)(BsT + (col0 + n) * LDB + k0 + kh * 16);
}

__device__ __forceinline__ v8f wmma_f16(v16h a, v16h b, v8f c) {
  return __builtin_amdgcn_wmma_f32_16x16x32_f16(false, a, false, b, (short)0, c,
                                                false, false);
}

// ---------------------------------------------------------------------------
// Convert weights f32 -> f16 into workspace. W_f1 (50x128) zero-padded to 64x128.
__global__ void __launch_bounds__(256)
prep_weights(const float* Wf1, const float* Wf2, const float* Win2f,
             const float* Wf2out, const float* Wdense, _Float16* ws16) {
  _Float16* W1h = ws16;                    // 64*128
  _Float16* W2h = W1h + 64 * Fdim;         // 128*128
  _Float16* Wih = W2h + Fdim * Fdim;
  _Float16* Woh = Wih + Fdim * Fdim;
  _Float16* Wdh = Woh + Fdim * Fdim;
  int i = blockIdx.x * blockDim.x + threadIdx.x;
  int stride = gridDim.x * blockDim.x;
  for (int j = i; j < 64 * Fdim; j += stride) {
    int g = j >> 7;
    W1h[j] = (g < Gdim) ? (_Float16)Wf1[j] : (_Float16)0.0f;
  }
  for (int j = i; j < Fdim * Fdim; j += stride) {
    W2h[j] = (_Float16)Wf2[j];
    Wih[j] = (_Float16)Win2f[j];
    Woh[j] = (_Float16)Wf2out[j];
    Wdh[j] = (_Float16)Wdense[j];
  }
}

// ---------------------------------------------------------------------------
// Row-tiled GEMM: out(Mx128) = act(A(Mx128) @ Wh(128x128) + bias).
// 64-row tile per block, 256 threads = 8 waves, each wave owns a 16x64 strip
// (rb = wave&3 row-block, cb0 = (wave>>2)*4 col-block base). act: 0=none 1=ssp
__global__ void __launch_bounds__(256)
gemm_rows(const float* __restrict__ A, const _Float16* __restrict__ Wh,
          const float* __restrict__ bias, float* __restrict__ out, int act) {
  __shared__ __align__(32) _Float16 As[64 * LDA2];
  __shared__ __align__(32) _Float16 BsT[Fdim * LDB];
  int tid = threadIdx.x;
  int row0 = blockIdx.x * 64;
  for (int i = tid; i < 64 * Fdim; i += 256) {
    int r = i >> 7, c = i & 127;
    As[r * LDA2 + c] = (_Float16)A[(size_t)(row0 + r) * Fdim + c];
  }
  for (int i = tid; i < Fdim * Fdim; i += 256) {
    int k = i >> 7, n = i & 127;
    BsT[n * LDB + k] = Wh[i];
  }
  __syncthreads();

  int wave = tid >> 5, lane = tid & 31;
  int rb = wave & 3, cb0 = (wave >> 2) * 4;
  v8f acc[4];
#pragma unroll
  for (int t = 0; t < 4; ++t) acc[t] = (v8f){0,0,0,0,0,0,0,0};
#pragma unroll
  for (int k0 = 0; k0 < Fdim; k0 += 32) {
    v16h a = frag_a(As, LDA2, rb * 16, k0);
#pragma unroll
    for (int t = 0; t < 4; ++t)
      acc[t] = wmma_f16(a, frag_b(BsT, (cb0 + t) * 16, k0), acc[t]);
  }
  int n16 = lane & 15, mh = lane >> 4;
#pragma unroll
  for (int t = 0; t < 4; ++t) {
    int f = (cb0 + t) * 16 + n16;
    float bv = bias ? bias[f] : 0.0f;
#pragma unroll
    for (int r = 0; r < 8; ++r) {
      float v = acc[t][r] + bv;
      if (act) v = sspf(v);
      out[(size_t)(row0 + rb * 16 + mh * 8 + r) * Fdim + f] = v;
    }
  }
}

// ---------------------------------------------------------------------------
// Fused per-atom CFConv core: one workgroup per (b,a).
//   C1 = ssp(fij(64x50->64) @ W1(64x128) + b1)
//   Wf = (C1 @ W2(128x128) + b2) * (cutoff(r)*mask) per row
//   agg[f] = sum_n Wf[n][f] * y[neighbor[n]][f]
// Cross-wave column reduction is done via per-(rb,mh) partial slots in LDS and
// a fixed-order final sum -> bit-exact deterministic across replays (no float
// atomics).
__global__ void __launch_bounds__(256)
cfconv_atom(const float* __restrict__ fij, const float* __restrict__ rij,
            const int* __restrict__ nbrs, const float* __restrict__ mask,
            const _Float16* __restrict__ W1h, const _Float16* __restrict__ W2h,
            const float* __restrict__ b1, const float* __restrict__ b2,
            const float* __restrict__ yws, float* __restrict__ aggws) {
  extern __shared__ v16h smem_v[];              // guarantees 32B alignment
  _Float16* As1 = (_Float16*)smem_v;            // 64*LDA1
  _Float16* BsT = As1 + 64 * LDA1;              // 128*LDB
  _Float16* A2  = BsT + Fdim * LDB;             // 64*LDA2
  float*    Ys  = (float*)(A2 + 64 * LDA2);     // 64*LDY
  float*    wrow = Ys + 64 * LDY;               // 64
  float*    partials = wrow + 64;               // 8*128

  int tid = threadIdx.x;
  int atom = blockIdx.x;                         // b*A + a
  int b = atom >> 9;                             // A = 512

  // stage-1 A: fij tile (64 x 50), K-padded to 64 with zeros
  for (int i = tid; i < 64 * 64; i += 256) {
    int n = i >> 6, g = i & 63;
    As1[n * LDA1 + g] =
        (g < Gdim) ? (_Float16)fij[((size_t)atom * Ndim + n) * Gdim + g]
                   : (_Float16)0.0f;
  }
  // W1 -> LDS column-major
  for (int i = tid; i < 64 * Fdim; i += 256) {
    int k = i >> 7, n = i & 127;
    BsT[n * LDB + k] = W1h[i];
  }
  if (tid < Ndim) {
    float r = rij[(size_t)atom * Ndim + tid];
    float m = mask[(size_t)atom * Ndim + tid];
    wrow[tid] = m * cutoff_w(r);
  }
  // gather y[neighbor] rows (f32, float4 loads)
  for (int i = tid; i < 64 * 32; i += 256) {
    int n = i >> 5, q = i & 31;
    int nb = nbrs[(size_t)atom * Ndim + n];
    float4 v = *(const float4*)(yws + ((size_t)(b * Adim + nb)) * Fdim + q * 4);
    float* dst = Ys + n * LDY + q * 4;
    dst[0] = v.x; dst[1] = v.y; dst[2] = v.z; dst[3] = v.w;
  }
  __syncthreads();

  int wave = tid >> 5, lane = tid & 31;
  int rb = wave & 3, cb0 = (wave >> 2) * 4;
  int n16 = lane & 15, mh = lane >> 4;

  // Stage 1: K = 64
  v8f acc[4];
#pragma unroll
  for (int t = 0; t < 4; ++t) acc[t] = (v8f){0,0,0,0,0,0,0,0};
#pragma unroll
  for (int k0 = 0; k0 < 64; k0 += 32) {
    v16h a = frag_a(As1, LDA1, rb * 16, k0);
#pragma unroll
    for (int t = 0; t < 4; ++t)
      acc[t] = wmma_f16(a, frag_b(BsT, (cb0 + t) * 16, k0), acc[t]);
  }
  // bias + ssp -> A2 (f16) for stage 2
#pragma unroll
  for (int t = 0; t < 4; ++t) {
    int f = (cb0 + t) * 16 + n16;
    float bv = b1[f];
#pragma unroll
    for (int r = 0; r < 8; ++r) {
      int row = rb * 16 + mh * 8 + r;
      A2[row * LDA2 + f] = (_Float16)sspf(acc[t][r] + bv);
    }
  }
  __syncthreads();                               // stage-1 reads of BsT done
  // W2 -> LDS column-major (overwrite)
  for (int i = tid; i < Fdim * Fdim; i += 256) {
    int k = i >> 7, n = i & 127;
    BsT[n * LDB + k] = W2h[i];
  }
  __syncthreads();

  // Stage 2: K = 128
  v8f acc2[4];
#pragma unroll
  for (int t = 0; t < 4; ++t) acc2[t] = (v8f){0,0,0,0,0,0,0,0};
#pragma unroll
  for (int k0 = 0; k0 < Fdim; k0 += 32) {
    v16h a = frag_a(A2, LDA2, rb * 16, k0);
#pragma unroll
    for (int t = 0; t < 4; ++t)
      acc2[t] = wmma_f16(a, frag_b(BsT, (cb0 + t) * 16, k0), acc2[t]);
  }
  // Hadamard with gathered y + weighted partial row-sums.
  // Each column f gets exactly 8 partials, slot s = rb*2+mh (unique writer).
  int s = rb * 2 + mh;
#pragma unroll
  for (int t = 0; t < 4; ++t) {
    int f = (cb0 + t) * 16 + n16;
    float b2v = b2[f];
    float p = 0.0f;
#pragma unroll
    for (int r = 0; r < 8; ++r) {
      int row = rb * 16 + mh * 8 + r;
      p += (acc2[t][r] + b2v) * wrow[row] * Ys[row * LDY + f];
    }
    partials[s * Fdim + f] = p;
  }
  __syncthreads();
  // deterministic fixed-order final reduction
  if (tid < Fdim) {
    float a = 0.0f;
#pragma unroll
    for (int q = 0; q < 8; ++q) a += partials[q * Fdim + tid];
    aggws[(size_t)atom * Fdim + tid] = a;
  }
}

// ---------------------------------------------------------------------------
extern "C" void kernel_launch(void* const* d_in, const int* in_sizes, int n_in,
                              void* d_out, int out_size, void* d_ws, size_t ws_size,
                              hipStream_t stream) {
  const float* x        = (const float*)d_in[0];
  const float* r_ij     = (const float*)d_in[1];
  const int*   neighbors= (const int*)d_in[2];
  const float* nmask    = (const float*)d_in[3];
  const float* f_ij     = (const float*)d_in[4];
  const float* W_in2f   = (const float*)d_in[5];
  const float* W_f1     = (const float*)d_in[6];
  const float* b_f1     = (const float*)d_in[7];
  const float* W_f2     = (const float*)d_in[8];
  const float* b_f2     = (const float*)d_in[9];
  const float* W_f2out  = (const float*)d_in[10];
  const float* b_f2out  = (const float*)d_in[11];
  const float* W_dense  = (const float*)d_in[12];
  const float* b_dense  = (const float*)d_in[13];

  // workspace layout: f16 weights then f32 buffers
  _Float16* W1h = (_Float16*)d_ws;          // 64*128
  _Float16* W2h = W1h + 64 * Fdim;          // 128*128
  _Float16* Wih = W2h + Fdim * Fdim;
  _Float16* Woh = Wih + Fdim * Fdim;
  _Float16* Wdh = Woh + Fdim * Fdim;
  float* y_ws   = (float*)(Wdh + Fdim * Fdim);   // 4096*128
  float* agg_ws = y_ws + (size_t)NROWS * Fdim;   // 4096*128 (reused in place)

  size_t smem = (size_t)(64 * LDA1 + Fdim * LDB + 64 * LDA2) * sizeof(_Float16) +
                (size_t)(64 * LDY + 64 + 8 * Fdim) * sizeof(float);   // ~99 KB

  prep_weights<<<32, 256, 0, stream>>>(W_f1, W_f2, W_in2f, W_f2out, W_dense, W1h);
  gemm_rows<<<NROWS / 64, 256, 0, stream>>>(x, Wih, nullptr, y_ws, 0);
  cfconv_atom<<<NROWS, 256, smem, stream>>>(f_ij, r_ij, neighbors, nmask,
                                            W1h, W2h, b_f1, b_f2, y_ws, agg_ws);
  gemm_rows<<<NROWS / 64, 256, 0, stream>>>(agg_ws, Woh, b_f2out, agg_ws, 1);
  gemm_rows<<<NROWS / 64, 256, 0, stream>>>(agg_ws, Wdh, b_dense, (float*)d_out, 0);
}
